// MaskEstimator_28063316312768
// MI455X (gfx1250) — matrix operand order repaired and checked by reading, not compiled, not run on hardware
//
#include <hip/hip_runtime.h>
#include <cstdint>
#include <cstddef>

// ---- CDNA5 WMMA types ----
typedef __attribute__((ext_vector_type(16))) __bf16 v16bf;
typedef __attribute__((ext_vector_type(8)))  float  v8f;

#define LDS_XH_STRIDE 520   // 512 + 8 pad (bf16 elements) -> 1040B rows, 16B aligned
#define LDS_W_STRIDE   40   // 32 + 8 pad  (bf16 elements) ->   80B rows, 16B aligned
#define WBUF_ELEMS   (512 * LDS_W_STRIDE)   // one weight staging buffer

__device__ __forceinline__ __bf16 f2bf(float f) { return (__bf16)f; }   // native v_cvt
__device__ __forceinline__ float  bf2f(__bf16 b) { return (float)b; }

union Frag { float4 f4[2]; v16bf v; };

// 16-element bf16 fragment via two ds_load_b128; second half at +gap elements.
// A (16x32): lane holds k = koff+0..15 contiguous     -> gap = 8
// B (32x16): lane holds k = kb+0..15 contiguous       -> gap = 8
__device__ __forceinline__ v16bf fragGap(const __bf16* p, int gap) {
  Frag f;
  f.f4[0] = *(const float4*)(p);
  f.f4[1] = *(const float4*)(p + gap);
  return f.v;
}

__device__ __forceinline__ v8f wmma_bf16(v16bf a, v16bf b, v8f c) {
  return __builtin_amdgcn_wmma_f32_16x16x32_bf16(false, a, false, b, (short)0, c, false, false);
}

#define LOG2E 1.4426950408889634f

// Branchless GLU: tanh(a) * sigmoid(g), via v_exp_f32 / v_rcp_f32 only.
__device__ __forceinline__ float glu_tanh(float a, float g) {
  float ta = 1.0f - 2.0f * __builtin_amdgcn_rcpf(1.0f + __builtin_amdgcn_exp2f((2.0f * LOG2E) * a));
  float sg = __builtin_amdgcn_rcpf(1.0f + __builtin_amdgcn_exp2f(-LOG2E * g));
  return ta * sg;
}

// Fused: h = glu(X @ W0[n] + b0[n]); out_band = glu(h @ W1[:,off:off+2DI] + b1[..])
// One workgroup: 64 rows x one band. 256 threads = 8 waves.
template<int DI>
__global__ __launch_bounds__(256)
void fused_band_kernel(const float* __restrict__ x,  const float* __restrict__ W0,
                       const float* __restrict__ b0, const float* __restrict__ W1,
                       const float* __restrict__ b1, float* __restrict__ out, int n0)
{
  extern __shared__ char smem[];
  __bf16* ldsX = (__bf16*)smem;                       // 64 x 520 bf16 (X tile)
  __bf16* ldsH = ldsX + 64 * LDS_XH_STRIDE;           // 64 x 520 bf16 (GLU(h))
  __bf16* ldsW = ldsH + 64 * LDS_XH_STRIDE;           // 2 x (512 x 40) bf16, double buffer
  __bf16* ldsP = ldsX;                                // stage-2 pre-act reuses X region

  const int tid  = threadIdx.x;
  const int wv   = tid >> 5;
  const int lane = tid & 31;
  const int nl   = lane & 15;
  const int hb   = lane >> 4;

  const int n    = n0 + (int)blockIdx.y;
  const int row0 = (int)blockIdx.x * 64;

  const int offo = 32 * ((DI / 8) - 1) + (int)blockIdx.y * DI;
  const int off2 = 2 * offo;

  // ---- Phase 0: stage X tile (64 rows x 512 K) as bf16 into LDS ----
  #pragma unroll
  for (int i = 0; i < 32; ++i) {
    int idx = tid + i * 256;
    int r   = idx >> 7;
    int c4  = (idx & 127) * 4;
    const float4 xv = *(const float4*)(x + (((size_t)(row0 + r) * 24 + n) << 9) + c4);
    union { __bf16 b[4]; uint2 u; } pk;
    pk.b[0] = f2bf(xv.x); pk.b[1] = f2bf(xv.y); pk.b[2] = f2bf(xv.z); pk.b[3] = f2bf(xv.w);
    *(uint2*)(ldsX + r * LDS_XH_STRIDE + c4) = pk.u;
  }

  // ---- Stage 1 ----
  const int mt  = wv >> 1;
  const int sub = wv & 1;

  // stage W0 slice: 32 K x (64 a-cols + 64 gate-cols), transposed [col][k]
  auto stageW0 = [&](int cb, int kt, __bf16* wb) {
    #pragma unroll
    for (int i = 0; i < 4; ++i) {
      int e  = tid + i * 256;
      int k  = e >> 5;
      int c4 = (e & 31) * 4;
      int gc = (c4 < 64) ? (cb * 64 + c4) : (512 + cb * 64 + (c4 - 64));
      const float4 w4 = *(const float4*)(W0 + (((size_t)n * 512 + kt * 32 + k) << 10) + gc);
      wb[(c4 + 0) * LDS_W_STRIDE + k] = f2bf(w4.x);
      wb[(c4 + 1) * LDS_W_STRIDE + k] = f2bf(w4.y);
      wb[(c4 + 2) * LDS_W_STRIDE + k] = f2bf(w4.z);
      wb[(c4 + 3) * LDS_W_STRIDE + k] = f2bf(w4.w);
    }
  };

  for (int cb = 0; cb < 8; ++cb) {
    v8f accA0 = {0,0,0,0,0,0,0,0}, accA1 = {0,0,0,0,0,0,0,0};
    v8f accG0 = {0,0,0,0,0,0,0,0}, accG1 = {0,0,0,0,0,0,0,0};
    stageW0(cb, 0, ldsW);                      // prologue into buffer 0
    for (int kt = 0; kt < 16; ++kt) {
      __syncthreads();                         // buffer (kt&1) ready
      if (kt < 15) stageW0(cb, kt + 1, ldsW + ((kt + 1) & 1) * WBUF_ELEMS);
      const __bf16* wb = ldsW + (kt & 1) * WBUF_ELEMS;

      const int m    = mt * 16 + nl;
      const int koff = kt * 32 + hb * 16;
      v16bf af = fragGap(ldsX + m * LDS_XH_STRIDE + koff, 8);
      const int kb = hb * 16;
      v16bf ba0 = fragGap(wb + (     sub * 32 +  0 + nl) * LDS_W_STRIDE + kb, 8);
      v16bf ba1 = fragGap(wb + (     sub * 32 + 16 + nl) * LDS_W_STRIDE + kb, 8);
      v16bf bg0 = fragGap(wb + (64 + sub * 32 +  0 + nl) * LDS_W_STRIDE + kb, 8);
      v16bf bg1 = fragGap(wb + (64 + sub * 32 + 16 + nl) * LDS_W_STRIDE + kb, 8);
      accA0 = wmma_bf16(af, ba0, accA0);
      accA1 = wmma_bf16(af, ba1, accA1);
      accG0 = wmma_bf16(af, bg0, accG0);
      accG1 = wmma_bf16(af, bg1, accG1);
    }
    // bias + GLU -> ldsH (bf16)
    #pragma unroll
    for (int j = 0; j < 2; ++j) {
      int gcol = cb * 64 + sub * 32 + j * 16 + nl;
      float ba = b0[n * 1024 + gcol];
      float bg = b0[n * 1024 + 512 + gcol];
      v8f aa = j ? accA1 : accA0;
      v8f gg = j ? accG1 : accG0;
      #pragma unroll
      for (int r = 0; r < 8; ++r)
        ldsH[(mt * 16 + hb * 8 + r) * LDS_XH_STRIDE + gcol] = f2bf(glu_tanh(aa[r] + ba, gg[r] + bg));
    }
  }

  // ---- Stage 2 ----
  constexpr int NTC    = (2 * DI) / 16;
  constexpr int NTILES = 4 * NTC;
  constexpr int TPW    = (NTILES + 7) / 8;
  v8f acc2[TPW];
  #pragma unroll
  for (int tt = 0; tt < TPW; ++tt) acc2[tt] = (v8f){0,0,0,0,0,0,0,0};

  // stage W1 slice: 32 K x 2*DI cols, transposed [col][k]
  auto stageW1 = [&](int kt, __bf16* wb) {
    for (int e = tid; e < 16 * DI; e += 256) {
      int k  = e / (DI / 2);
      int c4 = (e % (DI / 2)) * 4;
      const float4 w4 = *(const float4*)(W1 + (size_t)(kt * 32 + k) * 4032 + off2 + c4);
      wb[(c4 + 0) * LDS_W_STRIDE + k] = f2bf(w4.x);
      wb[(c4 + 1) * LDS_W_STRIDE + k] = f2bf(w4.y);
      wb[(c4 + 2) * LDS_W_STRIDE + k] = f2bf(w4.z);
      wb[(c4 + 3) * LDS_W_STRIDE + k] = f2bf(w4.w);
    }
  };

  stageW1(0, ldsW);
  for (int kt = 0; kt < 16; ++kt) {
    __syncthreads();
    if (kt < 15) stageW1(kt + 1, ldsW + ((kt + 1) & 1) * WBUF_ELEMS);
    const __bf16* wb = ldsW + (kt & 1) * WBUF_ELEMS;
    #pragma unroll
    for (int tt = 0; tt < TPW; ++tt) {
      int t = wv + tt * 8;                    // wave-uniform predicate
      if (t < NTILES) {
        int mt2 = t / NTC, ct = t % NTC;
        v16bf af = fragGap(ldsH + (mt2 * 16 + nl) * LDS_XH_STRIDE + kt * 32 + hb * 16, 8);
        v16bf bf = fragGap(wb + (ct * 16 + nl) * LDS_W_STRIDE + hb * 16, 8);
        acc2[tt] = wmma_bf16(af, bf, acc2[tt]);
      }
    }
  }

  // park pre-activations (bf16) in reused X region
  #pragma unroll
  for (int tt = 0; tt < TPW; ++tt) {
    int t = wv + tt * 8;
    if (t < NTILES) {
      int mt2 = t / NTC, ct = t % NTC;
      #pragma unroll
      for (int r = 0; r < 8; ++r)
        ldsP[(mt2 * 16 + hb * 8 + r) * LDS_XH_STRIDE + ct * 16 + nl] = f2bf(acc2[tt][r]);
    }
  }
  __syncthreads();

  // bias + GLU + coalesced store
  for (int idx = tid; idx < 64 * DI; idx += 256) {
    int m2 = idx / DI;
    int c  = idx % DI;
    float a = bf2f(ldsP[m2 * LDS_XH_STRIDE + c])      + b1[off2 + c];
    float g = bf2f(ldsP[m2 * LDS_XH_STRIDE + DI + c]) + b1[off2 + DI + c];
    out[(size_t)(row0 + m2) * 2016 + offo + c] = glu_tanh(a, g);
  }
}

extern "C" void kernel_launch(void* const* d_in, const int* in_sizes, int n_in,
                              void* d_out, int out_size, void* d_ws, size_t ws_size,
                              hipStream_t stream) {
  (void)in_sizes; (void)n_in; (void)out_size; (void)d_ws; (void)ws_size;
  const float* x  = (const float*)d_in[0];
  const float* W0 = (const float*)d_in[1];
  const float* b0 = (const float*)d_in[2];
  const float* W1 = (const float*)d_in[3];
  const float* b1 = (const float*)d_in[4];
  float* out = (float*)d_out;

  dim3 block(256);
  dim3 grid(32, 4);
  size_t shmem = (size_t)(2 * 64 * LDS_XH_STRIDE + 2 * WBUF_ELEMS) * 2;  // ~210 KB

  fused_band_kernel<  8><<<grid, block, shmem, stream>>>(x, W0, b0, W1, b1, out,  0);
  fused_band_kernel< 16><<<grid, block, shmem, stream>>>(x, W0, b0, W1, b1, out,  4);
  fused_band_kernel< 32><<<grid, block, shmem, stream>>>(x, W0, b0, W1, b1, out,  8);
  fused_band_kernel< 64><<<grid, block, shmem, stream>>>(x, W0, b0, W1, b1, out, 12);
  fused_band_kernel<128><<<grid, block, shmem, stream>>>(x, W0, b0, W1, b1, out, 16);
  fused_band_kernel<256><<<grid, block, shmem, stream>>>(x, W0, b0, W1, b1, out, 20);
}
